// GAT_21028159881586
// MI455X (gfx1250) — compile-verified, matrix operand order checked by reading, not run on hardware
//
#include <hip/hip_runtime.h>
#include <hip/hip_bf16.h>
#include <math.h>

typedef __attribute__((ext_vector_type(16))) _Float16 v16h;
typedef __attribute__((ext_vector_type(8)))  _Float16 v8h;
typedef __attribute__((ext_vector_type(8)))  float    v8f;
typedef __attribute__((ext_vector_type(4)))  float    f32x4;
typedef f32x4 f32x4u __attribute__((aligned(4)));   // unaligned-capable view

#define GAT_N   30000
#define GAT_E   480000
#define GAT_B   4096
#define GAT_IN  1281
#define GAT_HID 256

#define BK_MAX  1312          // max k rounded up to 32 (1281 -> 1312)
#define BSTRIDE 1320          // LDS column stride in halves: 660 dwords -> 16 cols
                              // hit 16 distinct banks (c*20 mod 64 all distinct)

__device__ __forceinline__ unsigned f32_key(float x) {
  unsigned u = __float_as_uint(x);
  return (u & 0x80000000u) ? ~u : (u | 0x80000000u);
}
__device__ __forceinline__ float f32_unkey(unsigned k) {
  unsigned u = (k & 0x80000000u) ? (k ^ 0x80000000u) : ~k;
  return __uint_as_float(u);
}
__device__ __forceinline__ void atomic_add_f32(float* p, float v) {
  __hip_atomic_fetch_add(p, v, __ATOMIC_RELAXED, __HIP_MEMORY_SCOPE_AGENT);
}

// ---------------------------------------------------------------------------
// WMMA GEMM: Y(n x m) = act( X(n x k) @ W(k x m) + bias )
// Block = 256 threads = 8 waves; block tile = 128 rows x 16 cols.
// B panel (k x 16) staged once in LDS as f16, column-major, zero-padded to
// a multiple of 32 so the K loop is completely branch-free.
// Per slab/wave: 4x global_load_b128 (A), 2x ds_load_b128 (B), 1x WMMA.
// act: 0 = none, 1 = relu
// ---------------------------------------------------------------------------
__global__ void __launch_bounds__(256)
gat_wmma_gemm(const float* __restrict__ X, const float* __restrict__ W,
              const float* __restrict__ bias, float* __restrict__ Y,
              int n, int k, int m, int act) {
  __shared__ __align__(16) _Float16 Bt[16 * BSTRIDE];

  const int wave  = threadIdx.x >> 5;
  const int lane  = threadIdx.x & 31;
  const int half  = lane >> 4;          // 0: lanes 0-15, 1: lanes 16-31
  const int l16   = lane & 15;
  const int tileM = blockIdx.x * 128 + wave * 16;
  const int tileN = blockIdx.y * 16;
  const int kpad  = (k + 31) & ~31;

  // ---- cooperative B panel load: W[k][16 cols] -> f16 LDS, zero padded ----
  for (int e = threadIdx.x; e < (kpad << 4); e += 256) {
    const int col = e & 15, k0 = e >> 4;
    const float v = (k0 < k) ? W[(size_t)k0 * m + tileN + col] : 0.f;
    Bt[col * BSTRIDE + k0] = (_Float16)v;
  }
  __syncthreads();

  const int rowc = min(tileM + l16, n - 1);       // clamped load row
  const float* __restrict__ xrow = X + (size_t)rowc * k;
  const _Float16* __restrict__ bcol = Bt + l16 * BSTRIDE + half * 8;
  const int h8 = half * 8;

  v8f c = {};
  const int kmain = k & ~31;

#pragma unroll 2
  for (int kk = 0; kk < kmain; kk += 32) {
    // A: two contiguous 8-float runs -> 4x b128 (align-4 safe, k may be odd)
    const f32x4u* p0 = (const f32x4u*)(xrow + kk + h8);
    const f32x4u* p1 = (const f32x4u*)(xrow + kk + 16 + h8);
    const f32x4 a0 = p0[0], a1 = p0[1], a2 = p1[0], a3 = p1[1];
    v16h a;
#pragma unroll
    for (int t = 0; t < 4; ++t) {
      a[t]      = (_Float16)a0[t];
      a[4 + t]  = (_Float16)a1[t];
      a[8 + t]  = (_Float16)a2[t];
      a[12 + t] = (_Float16)a3[t];
    }
    // B: two ds_load_b128 from the staged panel
    const v8h b0 = *(const v8h*)(bcol + kk);
    const v8h b1 = *(const v8h*)(bcol + kk + 16);
    const v16h b = __builtin_shufflevector(b0, b1, 0, 1, 2, 3, 4, 5, 6, 7,
                                           8, 9, 10, 11, 12, 13, 14, 15);
    c = __builtin_amdgcn_wmma_f32_16x16x32_f16(false, a, false, b,
                                               (short)0, c, false, false);
  }

  if (kmain < kpad) {                    // single checked tail slab
    v16h a;
#pragma unroll
    for (int j = 0; j < 8; ++j) {
      const int k0 = kmain + ((j & 3) << 1) + h8 + ((j >> 2) << 4);
      a[2 * j]     = (k0 < k)     ? (_Float16)xrow[k0]     : (_Float16)0.f;
      a[2 * j + 1] = (k0 + 1 < k) ? (_Float16)xrow[k0 + 1] : (_Float16)0.f;
    }
    const v8h b0 = *(const v8h*)(bcol + kmain);
    const v8h b1 = *(const v8h*)(bcol + kmain + 16);
    const v16h b = __builtin_shufflevector(b0, b1, 0, 1, 2, 3, 4, 5, 6, 7,
                                           8, 9, 10, 11, 12, 13, 14, 15);
    c = __builtin_amdgcn_wmma_f32_16x16x32_f16(false, a, false, b,
                                               (short)0, c, false, false);
  }

  const int col = tileN + l16;
  const float bv = bias ? bias[col] : 0.f;
#pragma unroll
  for (int j = 0; j < 8; ++j) {
    const int r = tileM + j + (half << 3);   // C/D layout: VGPR j -> M=j+8*half
    if (r < n) {
      float v = c[j] + bv;
      if (act == 1) v = v > 0.f ? v : 0.f;
      Y[(size_t)r * m + col] = v;
    }
  }
}

// ---------------------------------------------------------------------------
// Edge pass 1: per-(edge,head) attention logit + segment max (wave per e,h)
// ---------------------------------------------------------------------------
__global__ void gat_edge_logit_max(const float* __restrict__ xl,
                                   const float* __restrict__ xr,
                                   const int* __restrict__ ei,
                                   const float* __restrict__ att,
                                   float* __restrict__ elog,
                                   unsigned* __restrict__ maxb,
                                   int E, int Nn, int H, int C) {
  const int w    = (int)(((size_t)blockIdx.x * blockDim.x + threadIdx.x) >> 5);
  const int lane = threadIdx.x & 31;
  const int total = (E + Nn) * H;
  if (w >= total) return;
  const int e = w / H, h = w - e * H;
  const int src = (e < E) ? ei[e]     : (e - E);
  const int dst = (e < E) ? ei[E + e] : (e - E);
  const int HC = H * C;
  const float* pl = xl + (size_t)src * HC + h * C;
  const float* pr = xr + (size_t)dst * HC + h * C;
  const float* pa = att + h * C;
  float acc = 0.f;
  for (int cc = lane; cc < C; cc += 32) {
    float mm = pl[cc] + pr[cc];
    mm = mm > 0.f ? mm : 0.2f * mm;      // leaky_relu(0.2)
    acc += mm * pa[cc];
  }
#pragma unroll
  for (int off = 16; off; off >>= 1) acc += __shfl_xor(acc, off, 32);
  if (lane == 0) {
    elog[(size_t)e * H + h] = acc;
    atomicMax(maxb + (size_t)dst * H + h, f32_key(acc));
  }
}

// ---------------------------------------------------------------------------
// Edge pass 2: ex = exp(logit - max[dst]); den[dst] += ex  (thread per e,h)
// ---------------------------------------------------------------------------
__global__ void gat_edge_exp_den(const int* __restrict__ ei,
                                 float* __restrict__ elog,
                                 const unsigned* __restrict__ maxb,
                                 float* __restrict__ den,
                                 int E, int Nn, int H) {
  const int t = blockIdx.x * blockDim.x + threadIdx.x;
  const int total = (E + Nn) * H;
  if (t >= total) return;
  const int e = t / H, h = t - e * H;
  const int dst = (e < E) ? ei[E + e] : (e - E);
  const float mx = f32_unkey(maxb[(size_t)dst * H + h]);
  const float ex = expf(elog[t] - mx);
  elog[t] = ex;
  atomic_add_f32(den + (size_t)dst * H + h, ex);
}

// ---------------------------------------------------------------------------
// Edge pass 3: acc[dst] += alpha * xl[src]   (wave per edge, lanes = channels)
// ---------------------------------------------------------------------------
__global__ void gat_edge_scatter(const float* __restrict__ xl,
                                 const int* __restrict__ ei,
                                 const float* __restrict__ elog,
                                 const float* __restrict__ den,
                                 float* __restrict__ acc,
                                 int E, int Nn, int H, int C) {
  const int w    = (int)(((size_t)blockIdx.x * blockDim.x + threadIdx.x) >> 5);
  const int lane = threadIdx.x & 31;
  const int total = E + Nn;
  if (w >= total) return;
  const int src = (w < E) ? ei[w]     : (w - E);
  const int dst = (w < E) ? ei[E + w] : (w - E);
  const int HC = H * C;
  for (int ch = lane; ch < HC; ch += 32) {
    const int h = ch / C;
    const float alpha = elog[(size_t)w * H + h] /
                        (den[(size_t)dst * H + h] + 1e-16f);
    atomic_add_f32(acc + (size_t)dst * HC + ch,
                   xl[(size_t)src * HC + ch] * alpha);
  }
}

// ---------------------------------------------------------------------------
// Elementwise: out = elu(acc + bias[col])
// ---------------------------------------------------------------------------
__global__ void gat_bias_elu(const float* __restrict__ acc,
                             const float* __restrict__ bias,
                             float* __restrict__ outp, int n, int m) {
  const int t = blockIdx.x * blockDim.x + threadIdx.x;
  if (t >= n * m) return;
  const int cc = t % m;
  const float v = acc[t] + bias[cc];
  outp[t] = v > 0.f ? v : (expf(v) - 1.f);
}

// ---------------------------------------------------------------------------
// Build MLP input: [h2[idx[b]] | wt[b] | mut[b]]  (296 per row)
// ---------------------------------------------------------------------------
__global__ void gat_build_comb(const float* __restrict__ h2,
                               const int* __restrict__ idx,
                               const float* __restrict__ wt,
                               const float* __restrict__ mut,
                               float* __restrict__ comb, int B) {
  const int t = blockIdx.x * blockDim.x + threadIdx.x;
  if (t >= B * 296) return;
  const int b = t / 296, j = t - b * 296;
  float v;
  if (j < 256)      v = h2[(size_t)idx[b] * 256 + j];
  else if (j < 276) v = wt[b * 20 + (j - 256)];
  else              v = mut[b * 20 + (j - 276)];
  comb[t] = v;
}

// ---------------------------------------------------------------------------
// Final head: out[b] = z2[b,:] . hW3 + hb3   (wave per row, 64 channels)
// ---------------------------------------------------------------------------
__global__ void gat_final_dot(const float* __restrict__ z2,
                              const float* __restrict__ w,
                              const float* __restrict__ b,
                              float* __restrict__ out, int B) {
  const int wv   = (int)(((size_t)blockIdx.x * blockDim.x + threadIdx.x) >> 5);
  const int lane = threadIdx.x & 31;
  if (wv >= B) return;
  float acc = z2[(size_t)wv * 64 + lane] * w[lane] +
              z2[(size_t)wv * 64 + lane + 32] * w[lane + 32];
#pragma unroll
  for (int off = 16; off; off >>= 1) acc += __shfl_xor(acc, off, 32);
  if (lane == 0) out[wv] = acc + b[0];
}

// ---------------------------------------------------------------------------
extern "C" void kernel_launch(void* const* d_in, const int* in_sizes, int n_in,
                              void* d_out, int out_size, void* d_ws, size_t ws_size,
                              hipStream_t stream) {
  const float* x    = (const float*)d_in[0];
  const int*   ei   = (const int*)  d_in[1];
  const int*   vidx = (const int*)  d_in[2];
  const float* wt   = (const float*)d_in[3];
  const float* mut  = (const float*)d_in[4];
  const float* Wl1  = (const float*)d_in[5];
  const float* bl1  = (const float*)d_in[6];
  const float* Wr1  = (const float*)d_in[7];
  const float* br1  = (const float*)d_in[8];
  const float* att1 = (const float*)d_in[9];
  const float* bias1= (const float*)d_in[10];
  const float* Wl2  = (const float*)d_in[11];
  const float* bl2  = (const float*)d_in[12];
  const float* Wr2  = (const float*)d_in[13];
  const float* br2  = (const float*)d_in[14];
  const float* att2 = (const float*)d_in[15];
  const float* bias2= (const float*)d_in[16];
  const float* hW1  = (const float*)d_in[17];
  const float* hb1  = (const float*)d_in[18];
  const float* hW2  = (const float*)d_in[19];
  const float* hb2  = (const float*)d_in[20];
  const float* hW3  = (const float*)d_in[21];
  const float* hb3  = (const float*)d_in[22];
  float* out = (float*)d_out;

  const int Nn = GAT_N, E = GAT_E, B = GAT_B;
  const int ET = E + Nn;                     // edges incl. self loops

  // ---- workspace layout (floats) ----
  float* ws = (float*)d_ws;
  size_t o = 0;
  float*    xl   = ws + o; o += (size_t)Nn * 256;   // layer-l source transform
  float*    xr   = ws + o; o += (size_t)Nn * 256;   // layer-l target transform
  float*    hbuf = ws + o; o += (size_t)Nn * 256;   // aggregation acc / h
  float*    elog = ws + o; o += (size_t)ET * 4;     // edge logits->exp; MLP scratch
  float*    den  = ws + o; o += (size_t)Nn * 4;
  unsigned* maxb = (unsigned*)(ws + o); o += (size_t)Nn * 4;
  // MLP buffers overlap the (retired) edge-logit region
  float* comb = elog;                    // 4096*296
  float* z1   = comb + (size_t)B * 296;  // 4096*128
  float* z2   = z1   + (size_t)B * 128;  // 4096*64

  const dim3 blk(256);
  const dim3 gemmN((Nn + 127) / 128, 256 / 16);
  const int  ewaves1 = ((ET * 4 + 7) / 8);
  const int  ewaves2 = ((ET * 1 + 7) / 8);
  const int  escat   = ((ET + 7) / 8);

  // ===== GAT layer 1: xl = x@Wl1+bl1, xr = x@Wr1+br1 (k=1281) =====
  gat_wmma_gemm<<<gemmN, blk, 0, stream>>>(x, Wl1, bl1, xl, Nn, GAT_IN, 256, 0);
  gat_wmma_gemm<<<gemmN, blk, 0, stream>>>(x, Wr1, br1, xr, Nn, GAT_IN, 256, 0);

  hipMemsetAsync(maxb, 0, (size_t)Nn * 4 * sizeof(unsigned), stream);
  hipMemsetAsync(den,  0, (size_t)Nn * 4 * sizeof(float),    stream);
  hipMemsetAsync(hbuf, 0, (size_t)Nn * 256 * sizeof(float),  stream);

  gat_edge_logit_max<<<ewaves1, blk, 0, stream>>>(xl, xr, ei, att1, elog, maxb,
                                                  E, Nn, 4, 64);
  gat_edge_exp_den<<<(ET * 4 + 255) / 256, blk, 0, stream>>>(ei, elog, maxb, den,
                                                             E, Nn, 4);
  gat_edge_scatter<<<escat, blk, 0, stream>>>(xl, ei, elog, den, hbuf, E, Nn, 4, 64);
  gat_bias_elu<<<(Nn * 256 + 255) / 256, blk, 0, stream>>>(hbuf, bias1, hbuf, Nn, 256);

  // ===== GAT layer 2: xl = h1@Wl2+bl2, xr = h1@Wr2+br2 (k=256) =====
  gat_wmma_gemm<<<gemmN, blk, 0, stream>>>(hbuf, Wl2, bl2, xl, Nn, 256, 256, 0);
  gat_wmma_gemm<<<gemmN, blk, 0, stream>>>(hbuf, Wr2, br2, xr, Nn, 256, 256, 0);

  hipMemsetAsync(maxb, 0, (size_t)Nn * sizeof(unsigned), stream);
  hipMemsetAsync(den,  0, (size_t)Nn * sizeof(float),    stream);
  hipMemsetAsync(hbuf, 0, (size_t)Nn * 256 * sizeof(float), stream);

  gat_edge_logit_max<<<ewaves2, blk, 0, stream>>>(xl, xr, ei, att2, elog, maxb,
                                                  E, Nn, 1, 256);
  gat_edge_exp_den<<<(ET + 255) / 256, blk, 0, stream>>>(ei, elog, maxb, den,
                                                         E, Nn, 1);
  gat_edge_scatter<<<escat, blk, 0, stream>>>(xl, ei, elog, den, hbuf, E, Nn, 1, 256);
  gat_bias_elu<<<(Nn * 256 + 255) / 256, blk, 0, stream>>>(hbuf, bias2, hbuf, Nn, 256);

  // ===== MLP head =====
  gat_build_comb<<<(B * 296 + 255) / 256, blk, 0, stream>>>(hbuf, vidx, wt, mut,
                                                            comb, B);
  gat_wmma_gemm<<<dim3((B + 127) / 128, 128 / 16), blk, 0, stream>>>(
      comb, hW1, hb1, z1, B, 296, 128, 1);
  gat_wmma_gemm<<<dim3((B + 127) / 128, 64 / 16), blk, 0, stream>>>(
      z1, hW2, hb2, z2, B, 128, 64, 1);
  gat_final_dot<<<(B * 32 + 255) / 256, blk, 0, stream>>>(z2, hW3, hb3, out, B);

  (void)in_sizes; (void)n_in; (void)out_size; (void)ws_size;
}
